// Conv2d_NN_Attn_44908178047126
// MI455X (gfx1250) — compile-verified, matrix operand order checked by reading, not run on hardware
//
#include <hip/hip_runtime.h>

// ---------------------------------------------------------------------------
// Problem constants
// ---------------------------------------------------------------------------
#define BD 8       // batch
#define CD 64      // channels (in == out)
#define TD 4096    // tokens (H*W)
#define NK 9       // top-K neighbors / conv kernel width
#define KC (CD*NK) // 576 = conv GEMM K dim (multiple of 32)

#define GEMM_NP 256          // N columns per block (4 waves x 4 tiles x 16)
#define GEMM_KSTEP 32        // K per wmma

typedef __bf16  v16bf __attribute__((ext_vector_type(16)));
typedef float   v8f   __attribute__((ext_vector_type(8)));
typedef unsigned int u32x4 __attribute__((ext_vector_type(4)));
typedef int          i32x4 __attribute__((ext_vector_type(4)));
typedef int          i32x8 __attribute__((ext_vector_type(8)));

#if defined(__HIP_DEVICE_COMPILE__) && \
    __has_builtin(__builtin_amdgcn_tensor_load_to_lds) && \
    __has_builtin(__builtin_amdgcn_s_wait_tensorcnt)
#define USE_TDM 1
#else
#define USE_TDM 0
#endif

// ---------------------------------------------------------------------------
// WMMA operand loader (wave32, 16x16x32 bf16), row-major K-contiguous source.
// lane&15 selects the row (M or N), (lane>>4)*8 selects the K-half:
// elements [0..7] = K {kh..kh+7}, [8..15] = K {kh+16..kh+23}.
// ---------------------------------------------------------------------------
__device__ inline v16bf load_tile_rowmajor(const __bf16* base, int row, int ld,
                                           int kk, int lane) {
  int m  = lane & 15;
  int kh = (lane >> 4) * 8;
  const __bf16* p = base + (size_t)(row + m) * ld + kk + kh;
  v16bf r;
#pragma unroll
  for (int u = 0; u < 8; ++u) {
    r[u]     = p[u];
    r[u + 8] = p[u + 16];
  }
  return r;
}

#if USE_TDM
// ---------------------------------------------------------------------------
// Tensor Data Mover: stage a 2-D [rows x row_elems] bf16 tile (row stride =
// stride_elems) from global into LDS at byte offset lds_off.
// 6-arg builtin form (clang-23 / therock headers): groups 2,3 = NULL (2-D
// tensor case per ISA §7.3), extra i32x8 group zeroed, cpol immediate last.
// Descriptor layout per CDNA5 ISA 8.3/8.4 (group0: count/lds/global addr,
// group1: data_size=2B, tensor dims, tile dims, dim0 stride).
// ---------------------------------------------------------------------------
__device__ inline void tdm_load_2d(const __bf16* gsrc, unsigned lds_off,
                                   unsigned rows, unsigned row_elems,
                                   unsigned tensor_w, unsigned tensor_h,
                                   unsigned stride_elems) {
  unsigned long long ga = (unsigned long long)(const void*)gsrc;
  u32x4 g0;
  g0[0] = 1u;                                        // count=1 valid user D#
  g0[1] = lds_off;                                   // lds_addr (bytes)
  g0[2] = (unsigned)(ga & 0xFFFFFFFFu);              // global_addr[31:0]
  g0[3] = (unsigned)((ga >> 32) & 0x01FFFFFFu) | (2u << 30); // addr[56:32]|type=2
  i32x8 g1;
  g1[0] = (int)(1u << 16);                           // data_size=1 -> 2 bytes
  g1[1] = (int)((tensor_w & 0xFFFFu) << 16);         // tensor_dim0[15:0]
  g1[2] = (int)((tensor_w >> 16) | ((tensor_h & 0xFFFFu) << 16));
  g1[3] = (int)((tensor_h >> 16) | (row_elems << 16)); // tile_dim0
  g1[4] = (int)rows;                                 // tile_dim1 (tile_dim2=0)
  g1[5] = (int)stride_elems;                         // tensor_dim0_stride lo32
  g1[6] = 0;
  g1[7] = 0;
  i32x4 z4 = {0, 0, 0, 0};
  i32x8 z8 = {0, 0, 0, 0, 0, 0, 0, 0};
  __builtin_amdgcn_tensor_load_to_lds(g0, g1, z4, z4, z8, 0);
}
#endif

// ---------------------------------------------------------------------------
// fp32 -> bf16 conversion (x, Wq/Wk/Wv/Wo, conv_w)
// ---------------------------------------------------------------------------
__global__ void cvt_f32_bf16(const float* __restrict__ in,
                             __bf16* __restrict__ out, int n) {
  int i = blockIdx.x * blockDim.x + threadIdx.x;
  if (i < n) out[i] = (__bf16)in[i];
}

// ---------------------------------------------------------------------------
// GEMM: C[m][n] = sum_k A[m][k] * Bm[n][k]   (A:[M][Kd], Bm:[N][Kd], bf16)
// Block: 128 threads / 4 waves computing a 16(M) x 256(N) panel.
// B slab (256 x 32 bf16 = 16KB) staged in LDS by the Tensor Data Mover,
// double-buffered; handoff = s_wait_tensorcnt + workgroup barrier.
// Each wave holds 4 accumulators (16x64), one A fragment amortized 4x.
// Weights are L2-resident (bf16 32MB << 192MB L2) so re-reads never hit HBM.
// ---------------------------------------------------------------------------
__global__ void __launch_bounds__(128)
gemm_bf16_nt(const __bf16* __restrict__ A,
             const __bf16* __restrict__ Bm,
             float* __restrict__ C,
             int M, int N, int Kd) {
  __shared__ __bf16 sB[2][GEMM_NP * GEMM_KSTEP];   // 2 x 16 KB

  int tid  = threadIdx.x;
  int lane = tid & 31;
  int wave = tid >> 5;
  int rowBase = blockIdx.y * 16;
  int nBase   = blockIdx.x * GEMM_NP;

#if USE_TDM
  unsigned lds_base = (unsigned)(unsigned long long)(void*)&sB[0][0];
  const unsigned slabBytes = GEMM_NP * GEMM_KSTEP * 2;
  // prologue: DMA first B slab
  if (wave == 0) {
    tdm_load_2d(Bm + (size_t)nBase * Kd, lds_base,
                GEMM_NP, GEMM_KSTEP, (unsigned)Kd, (unsigned)(N - nBase),
                (unsigned)Kd);
    __builtin_amdgcn_s_wait_tensorcnt(0);
  }
#else
  for (int e = tid; e < GEMM_NP * GEMM_KSTEP; e += blockDim.x)
    sB[0][e] = Bm[(size_t)(nBase + (e >> 5)) * Kd + (e & 31)];
#endif
  __syncthreads();

  v8f acc0 = {}, acc1 = {}, acc2 = {}, acc3 = {};
  const __bf16* Arow = A + (size_t)rowBase * Kd;
  int p = 0;
  int r0 = wave * 64 + (lane & 15);
  int kh = (lane >> 4) * 8;

  for (int kk = 0; kk < Kd; kk += GEMM_KSTEP) {
    // kick off DMA of the next slab into the other buffer
    if (kk + GEMM_KSTEP < Kd) {
#if USE_TDM
      if (wave == 0) {
        tdm_load_2d(Bm + (size_t)nBase * Kd + (kk + GEMM_KSTEP),
                    lds_base + (unsigned)(p ^ 1) * slabBytes,
                    GEMM_NP, GEMM_KSTEP, (unsigned)Kd,
                    (unsigned)(N - nBase), (unsigned)Kd);
      }
#else
      for (int e = tid; e < GEMM_NP * GEMM_KSTEP; e += blockDim.x)
        sB[p ^ 1][e] =
            Bm[(size_t)(nBase + (e >> 5)) * Kd + (kk + GEMM_KSTEP) + (e & 31)];
#endif
    }

    if (kk + 64 < Kd)
      __builtin_prefetch(Arow + (size_t)(lane & 15) * Kd + kk + 64, 0, 0);

    v16bf a = load_tile_rowmajor(A, rowBase, Kd, kk, lane);
    const __bf16* sb = sB[p];

#pragma unroll
    for (int u = 0; u < 4; ++u) {
      const __bf16* q = sb + (size_t)(r0 + u * 16) * GEMM_KSTEP + kh;
      v16bf b;
#pragma unroll
      for (int e = 0; e < 8; ++e) { b[e] = q[e]; b[e + 8] = q[e + 16]; }
      switch (u) {
        case 0: acc0 = __builtin_amdgcn_wmma_f32_16x16x32_bf16(false, a, false, b, (short)0, acc0, false, false); break;
        case 1: acc1 = __builtin_amdgcn_wmma_f32_16x16x32_bf16(false, a, false, b, (short)0, acc1, false, false); break;
        case 2: acc2 = __builtin_amdgcn_wmma_f32_16x16x32_bf16(false, a, false, b, (short)0, acc2, false, false); break;
        case 3: acc3 = __builtin_amdgcn_wmma_f32_16x16x32_bf16(false, a, false, b, (short)0, acc3, false, false); break;
      }
    }

    if (kk + GEMM_KSTEP < Kd) {
#if USE_TDM
      if (wave == 0) __builtin_amdgcn_s_wait_tensorcnt(0);
#endif
      __syncthreads();
      p ^= 1;
    }
  }

  int n0 = lane & 15;
  int mo = (lane >> 4) * 8;
  float* cbase = C + (size_t)rowBase * N + nBase + wave * 64 + n0;
#pragma unroll
  for (int r = 0; r < 8; ++r) {
    cbase[(size_t)(mo + r) * N +  0] = acc0[r];
    cbase[(size_t)(mo + r) * N + 16] = acc1[r];
    cbase[(size_t)(mo + r) * N + 32] = acc2[r];
    cbase[(size_t)(mo + r) * N + 48] = acc3[r];
  }
}

// ---------------------------------------------------------------------------
// Channel-norm of q,k; transpose to [B*T][64] bf16 (WMMA K-contiguous).
// One wave per (b,t); lane covers channels {lane, lane+32}.
// ---------------------------------------------------------------------------
__global__ void normalize_qk(const float* __restrict__ q,
                             const float* __restrict__ k,
                             __bf16* __restrict__ qnT,
                             __bf16* __restrict__ knT) {
  int bt = blockIdx.x;            // 0 .. B*T-1
  int b  = bt >> 12;
  int t  = bt & (TD - 1);
  int lane = threadIdx.x & 31;
  int c0 = lane, c1 = lane + 32;

  float q0 = q[((size_t)b * CD + c0) * TD + t];
  float q1 = q[((size_t)b * CD + c1) * TD + t];
  float k0 = k[((size_t)b * CD + c0) * TD + t];
  float k1 = k[((size_t)b * CD + c1) * TD + t];

  float sq = q0 * q0 + q1 * q1;
  float sk = k0 * k0 + k1 * k1;
#pragma unroll
  for (int off = 16; off > 0; off >>= 1) {
    sq += __shfl_xor(sq, off, 32);
    sk += __shfl_xor(sk, off, 32);
  }
  float rq = 1.0f / fmaxf(sqrtf(sq), 1e-12f);
  float rk = 1.0f / fmaxf(sqrtf(sk), 1e-12f);

  __bf16* qp = qnT + (size_t)bt * CD;
  __bf16* kp = knT + (size_t)bt * CD;
  qp[c0] = (__bf16)(q0 * rq);
  qp[c1] = (__bf16)(q1 * rq);
  kp[c0] = (__bf16)(k0 * rk);
  kp[c1] = (__bf16)(k1 * rk);
}

// ---------------------------------------------------------------------------
// Similarity + top-9:  sim[b,t,s] = relu( kn[:,t] . qn[:,s] ).
// One wave per (b, 16-token tile); WMMA A=qnT rows (M=s), B=knT rows (N=t).
// Lane owns token t=lane&15; lanes L and L^16 split the s range, merged in LDS.
// ---------------------------------------------------------------------------
__global__ void sim_topk(const __bf16* __restrict__ qnT,
                         const __bf16* __restrict__ knT,
                         int* __restrict__ idx) {
  __shared__ float sv[32][NK];
  __shared__ int   si[32][NK];

  int blk   = blockIdx.x;            // 0 .. B*(T/16)-1
  int b     = blk >> 8;
  int tBase = (blk & 255) * 16;
  int lane  = threadIdx.x & 31;

  const __bf16* qb = qnT + (size_t)b * TD * CD;
  const __bf16* kb = knT + (size_t)b * TD * CD;

  v16bf bt0 = load_tile_rowmajor(kb, tBase, CD, 0,  lane);
  v16bf bt1 = load_tile_rowmajor(kb, tBase, CD, 32, lane);

  float tv[NK];
  int   ti[NK];
#pragma unroll
  for (int j = 0; j < NK; ++j) { tv[j] = -1.0f; ti[j] = 0; }

  int mo = (lane >> 4) * 8;
  for (int sTile = 0; sTile < TD / 16; ++sTile) {
    int sBase = sTile * 16;
    v16bf a0 = load_tile_rowmajor(qb, sBase, CD, 0,  lane);
    v16bf a1 = load_tile_rowmajor(qb, sBase, CD, 32, lane);
    v8f acc = {};
    acc = __builtin_amdgcn_wmma_f32_16x16x32_bf16(false, a0, false, bt0,
                                                  (short)0, acc, false, false);
    acc = __builtin_amdgcn_wmma_f32_16x16x32_bf16(false, a1, false, bt1,
                                                  (short)0, acc, false, false);
#pragma unroll
    for (int r = 0; r < 8; ++r) {
      float vsim = fmaxf(acc[r], 0.0f);
      int   s    = sBase + mo + r;
      if (vsim > tv[NK - 1]) {
        tv[NK - 1] = vsim; ti[NK - 1] = s;
#pragma unroll
        for (int p = NK - 1; p > 0; --p) {
          if (tv[p] > tv[p - 1]) {
            float fv = tv[p]; tv[p] = tv[p - 1]; tv[p - 1] = fv;
            int   fi = ti[p]; ti[p] = ti[p - 1]; ti[p - 1] = fi;
          }
        }
      }
    }
  }

#pragma unroll
  for (int j = 0; j < NK; ++j) { sv[lane][j] = tv[j]; si[lane][j] = ti[j]; }
  __syncthreads();

  if (lane < 16) {
    const float* av = sv[lane];      const int* ai = si[lane];
    const float* bv = sv[lane + 16]; const int* bi = si[lane + 16];
    int pa = 0, pb = 0;
    int* op = idx + ((size_t)b * TD + tBase + lane) * NK;
#pragma unroll
    for (int j = 0; j < NK; ++j) {
      bool ta = (av[pa] > bv[pb]) || (av[pa] == bv[pb] && ai[pa] < bi[pb]);
      if (ta) { op[j] = ai[pa]; ++pa; }
      else    { op[j] = bi[pb]; ++pb; }
    }
  }
}

// ---------------------------------------------------------------------------
// Gather neighbors + Conv1d(K=9, stride=9) as a WMMA GEMM per 16-token tile:
// out1[o,t] = bias[o] + sum_{k=i*9+j} cw[o][k] * v[i][idx[t][j]].
// Gathered 576x16 bf16 tile staged in LDS; 4 waves cover 64 out channels.
// ---------------------------------------------------------------------------
__global__ void gather_conv(const float* __restrict__ v,
                            const int* __restrict__ idx,
                            const __bf16* __restrict__ cwb,   // [64][576]
                            const float* __restrict__ bias,
                            __bf16* __restrict__ out1) {      // [B*64][T]
  __shared__ __bf16 sB[KC * 16];    // 18 KB

  int blk   = blockIdx.x;           // 0 .. B*(T/16)-1
  int b     = blk >> 8;
  int tBase = (blk & 255) * 16;
  int lane  = threadIdx.x & 31;
  int wave  = threadIdx.x >> 5;

  for (int e = threadIdx.x; e < KC * 16; e += blockDim.x) {
    int k = e >> 4;
    int n = e & 15;
    int i = k / NK;
    int j = k - i * NK;
    int s = idx[((size_t)b * TD + tBase + n) * NK + j];
    sB[e] = (__bf16)v[((size_t)b * CD + i) * TD + s];
  }
  __syncthreads();

  int oBase = wave * 16;
  v8f acc = {};
  int nn = lane & 15;
  int kh = (lane >> 4) * 8;
  for (int kk = 0; kk < KC; kk += 32) {
    v16bf a = load_tile_rowmajor(cwb, oBase, KC, kk, lane);
    v16bf bm;
#pragma unroll
    for (int u = 0; u < 8; ++u) {
      bm[u]     = sB[(kk + kh + u) * 16 + nn];
      bm[u + 8] = sB[(kk + kh + 16 + u) * 16 + nn];
    }
    acc = __builtin_amdgcn_wmma_f32_16x16x32_bf16(false, a, false, bm,
                                                  (short)0, acc, false, false);
  }

  int mo = (lane >> 4) * 8;
#pragma unroll
  for (int r = 0; r < 8; ++r) {
    int o = oBase + mo + r;
    out1[((size_t)b * CD + o) * TD + tBase + nn] = (__bf16)(acc[r] + bias[o]);
  }
}

// ---------------------------------------------------------------------------
// Host-side orchestration
// ---------------------------------------------------------------------------
extern "C" void kernel_launch(void* const* d_in, const int* in_sizes, int n_in,
                              void* d_out, int out_size, void* d_ws, size_t ws_size,
                              hipStream_t stream) {
  const float* x      = (const float*)d_in[0];
  const float* Wq     = (const float*)d_in[1];
  const float* Wk     = (const float*)d_in[2];
  const float* Wv     = (const float*)d_in[3];
  const float* Wo     = (const float*)d_in[4];
  const float* conv_w = (const float*)d_in[5];
  const float* conv_b = (const float*)d_in[6];
  float* out = (float*)d_out;

  const size_t M   = (size_t)BD * CD;       // 512
  const size_t NXF = M * TD;                // 2,097,152
  const size_t NW  = (size_t)TD * TD;       // 16,777,216
  const size_t NCW = (size_t)CD * CD * NK;  // 36,864

  char* ws = (char*)d_ws;
  size_t off = 0;
  auto carve = [&](size_t bytes) -> void* {
    void* p = ws + off;
    off += (bytes + 255) & ~(size_t)255;
    return p;
  };

  __bf16* xf_b  = (__bf16*)carve(NXF * 2);
  __bf16* wq_b  = (__bf16*)carve(NW * 2);
  __bf16* wk_b  = (__bf16*)carve(NW * 2);
  __bf16* wv_b  = (__bf16*)carve(NW * 2);
  __bf16* wo_b  = (__bf16*)carve(NW * 2);
  __bf16* cw_b  = (__bf16*)carve(NCW * 2);
  float*  qf    = (float*)carve(NXF * 4);
  float*  kf    = (float*)carve(NXF * 4);
  float*  vf    = (float*)carve(NXF * 4);
  __bf16* qnT   = (__bf16*)carve(NXF * 2);
  __bf16* knT   = (__bf16*)carve(NXF * 2);
  int*    idx   = (int*)carve((size_t)BD * TD * NK * 4);
  __bf16* out1  = (__bf16*)carve(NXF * 2);
  (void)ws_size; (void)in_sizes; (void)n_in; (void)out_size;

  // 1) convert to bf16
  cvt_f32_bf16<<<(int)((NXF + 255) / 256), 256, 0, stream>>>(x, xf_b, (int)NXF);
  cvt_f32_bf16<<<(int)((NW + 255) / 256),  256, 0, stream>>>(Wq, wq_b, (int)NW);
  cvt_f32_bf16<<<(int)((NW + 255) / 256),  256, 0, stream>>>(Wk, wk_b, (int)NW);
  cvt_f32_bf16<<<(int)((NW + 255) / 256),  256, 0, stream>>>(Wv, wv_b, (int)NW);
  cvt_f32_bf16<<<(int)((NW + 255) / 256),  256, 0, stream>>>(Wo, wo_b, (int)NW);
  cvt_f32_bf16<<<(int)((NCW + 255) / 256), 256, 0, stream>>>(conv_w, cw_b, (int)NCW);

  // 2) q/k/v projections (TDM-staged, double-buffered WMMA GEMM)
  dim3 ggrid(TD / GEMM_NP, (unsigned)(M / 16));
  gemm_bf16_nt<<<ggrid, 128, 0, stream>>>(xf_b, wq_b, qf, (int)M, TD, TD);
  gemm_bf16_nt<<<ggrid, 128, 0, stream>>>(xf_b, wk_b, kf, (int)M, TD, TD);
  gemm_bf16_nt<<<ggrid, 128, 0, stream>>>(xf_b, wv_b, vf, (int)M, TD, TD);

  // 3) cosine-normalize over channels, transpose to [B*T][64] bf16
  normalize_qk<<<BD * TD, 32, 0, stream>>>(qf, kf, qnT, knT);

  // 4) similarity tiles via WMMA + per-token top-9
  sim_topk<<<BD * (TD / 16), 32, 0, stream>>>(qnT, knT, idx);

  // 5) neighbor gather + Conv1d-as-GEMM (+bias)
  gather_conv<<<BD * (TD / 16), 128, 0, stream>>>(vf, idx, cw_b, conv_b, out1);

  // 6) output projection with Wo -> fp32 result
  gemm_bf16_nt<<<ggrid, 128, 0, stream>>>(out1, wo_b, out, (int)M, TD, TD);
}
